// EdgeApplyModule_31447750542169
// MI455X (gfx1250) — matrix operand order repaired and checked by reading, not compiled, not run on hardware
//
#include <hip/hip_runtime.h>

typedef __attribute__((ext_vector_type(16))) __bf16    v16bf;
typedef __attribute__((ext_vector_type(8)))  float     v8f;
typedef __attribute__((ext_vector_type(8)))  unsigned  v8u;
typedef __attribute__((ext_vector_type(4)))  float     vf4;

#define D_OUT    64
#define TILE_M   128          // 8 waves x 16 rows
#define KT       6            // 192 / 32
#define LDD      68           // fp32 elems per D row (64 + 4 pad -> conflict-free)
#define WPK_U32  (KT * 4 * 32 * 8)   // 6144 dwords = 24576 B packed-W image

// fp32 -> bf16 RNE (repack kernel; integer path)
__device__ __forceinline__ unsigned short f2bf(float f) {
    unsigned u = __builtin_bit_cast(unsigned, f);
    unsigned r = 0x7FFFu + ((u >> 16) & 1u);
    return (unsigned short)((u + r) >> 16);
}
__device__ __forceinline__ unsigned pack2_int(float a, float b) {
    return (unsigned)f2bf(a) | ((unsigned)f2bf(b) << 16);
}

// Hot-path fp32x2 -> packed bf16x2 via VOP3P mix converts
// (V_FMA_MIXLO_BF16 / V_FMA_MIXHI_BF16: D[15:0]/D[31:16] = cvt(fma(s0,1.0,0)))
__device__ __forceinline__ unsigned pack2(float a, float b) {
    unsigned d;
    asm("v_fma_mixlo_bf16 %0, %1, 1.0, 0\n\t"
        "v_fma_mixhi_bf16 %0, %2, 1.0, 0"
        : "=&v"(d) : "v"(a), "v"(b));
    return d;
}

// ---- One-shot: repack W (fp32 row-major [192][64]) into the exact WMMA
//      B-fragment bf16 image: u = ((kt*4 + nt)*32 + lane)*8 + r
//      B 32x16 layout: lane<16 -> N=lane,K=2r,2r+1 (r<4) / 8+2r (r>=4) ; lane>=16 -> +8
__global__ __launch_bounds__(256)
void pack_w_frags(const float* __restrict__ W, unsigned* __restrict__ wpk) {
    int u = blockIdx.x * 256 + threadIdx.x;
    if (u >= WPK_U32) return;
    int r    = u & 7;
    int lane = (u >> 3) & 31;
    int nt   = (u >> 8) & 3;
    int kt   = u >> 10;
    int n    = nt * 16 + (lane & 15);
    int k    = kt * 32 + ((lane >> 4) << 3) + 2 * r + ((r >> 2) << 3);
    wpk[u] = pack2_int(W[k * D_OUT + n], W[(k + 1) * D_OUT + n]);
}

__global__ __launch_bounds__(256)
void edge_apply_wmma_bf16(const float* __restrict__ node_feat,
                          const float* __restrict__ edge_feat,
                          const int*   __restrict__ src_idx,
                          const int*   __restrict__ dst_idx,
                          const unsigned* __restrict__ wpk,
                          const float* __restrict__ bias,
                          float*       __restrict__ out,
                          int E)
{
    // [0, 24576)    : shB - packed W fragments (bf16, WMMA B layout)
    // [24576, ...)  : shD - fp32 D tile staging (128 x 68)
    __shared__ __align__(32) unsigned char smem[24576 + TILE_M * LDD * 4];
    unsigned* shB = (unsigned*)smem;
    float*    shD = (float*)(smem + 24576);

    const int tid = threadIdx.x;
    const int e0  = blockIdx.x * TILE_M;

    // ---- Async copy packed W image -> LDS (96 B per thread, 6 x b128) ----
    {
        unsigned ldsb = ((unsigned)(size_t)(void*)shB) + tid * 16;  // low 32 = LDS offset
        unsigned voff = tid * 16;
        unsigned long long gbase = (unsigned long long)(size_t)wpk;
        #pragma unroll
        for (int i = 0; i < 6; ++i) {
            asm volatile("global_load_async_to_lds_b128 %0, %1, %2"
                         :: "v"(ldsb), "v"(voff), "s"(gbase) : "memory");
            ldsb += 4096;
            voff += 4096;
        }
    }

    // ---- Per-wave geometry: 16 rows x full 64 output cols ----
    const int wave = tid >> 5;
    const int lane = tid & 31;
    const bool low = lane < 16;
    const int  ln  = lane & 15;
    const int  koff = low ? 0 : 8;          // A-fragment K sub-offset per lane half

    // Row for this lane (both lane halves use the same row)
    const int r  = e0 + wave * 16 + ln;
    const int rc = (r < E) ? r : (E - 1);

    // Concat segments: K 0-63 = node[src] (L2-hot), 64-127 = edge (streamed),
    //                  K 128-191 = node[dst] (L2-hot)
    const float* sp = node_feat + (size_t)src_idx[rc] * 64;
    const float* ep = edge_feat + (size_t)rc          * 64;
    const float* dp = node_feat + (size_t)dst_idx[rc] * 64;
    const float* segs[3] = { sp, ep, dp };

    asm volatile("s_wait_asynccnt 0x0" ::: "memory");   // W image resident in LDS
    __syncthreads();

    v8f acc[4] = {};

    #pragma unroll
    for (int kt = 0; kt < KT; ++kt) {
        // A fragment (16x32 bf16): lanes 0-15: K kb..kb+7 (V0-3), kb+16..23 (V4-7);
        // lanes 16-31: +8. Built directly from global fp32 with mix-converts.
        const float* base = segs[kt >> 1] + (kt & 1) * 32 + koff;
        vf4 f0, f1, f2, f3;
        if ((kt >> 1) == 1) {   // edge_feat: read-once stream -> non-temporal
            f0 = __builtin_nontemporal_load((const vf4*)(base));
            f1 = __builtin_nontemporal_load((const vf4*)(base + 4));
            f2 = __builtin_nontemporal_load((const vf4*)(base + 16));
            f3 = __builtin_nontemporal_load((const vf4*)(base + 20));
        } else {                // node_feat gathers: keep L2-resident
            f0 = *(const vf4*)(base);
            f1 = *(const vf4*)(base + 4);
            f2 = *(const vf4*)(base + 16);
            f3 = *(const vf4*)(base + 20);
        }
        v8u av;
        av[0] = pack2(f0.x, f0.y); av[1] = pack2(f0.z, f0.w);
        av[2] = pack2(f1.x, f1.y); av[3] = pack2(f1.z, f1.w);
        av[4] = pack2(f2.x, f2.y); av[5] = pack2(f2.z, f2.w);
        av[6] = pack2(f3.x, f3.y); av[7] = pack2(f3.z, f3.w);
        v16bf a = __builtin_bit_cast(v16bf, av);

        #pragma unroll
        for (int nt = 0; nt < 4; ++nt) {
            v16bf b = __builtin_bit_cast(v16bf,
                          *(const v8u*)(shB + ((kt * 4 + nt) * 32 + lane) * 8));
            acc[nt] = __builtin_amdgcn_wmma_f32_16x16x32_bf16(
                          false, a, false, b, (short)0, acc[nt], false, false);
        }
    }

    // ---- Bias + ReLU, stage D tile in LDS (conflict-free layout) ----
    // D layout: VGPR g -> row m0+g (lanes 0-15) / m0+g+8 (lanes 16-31), col = nt*16+ln
    {
        const int rb = wave * 16 + (low ? 0 : 8);
        #pragma unroll
        for (int nt = 0; nt < 4; ++nt) {
            const float bv = bias[nt * 16 + ln];
            const int   c  = nt * 16 + ln;
            #pragma unroll
            for (int g = 0; g < 8; ++g) {
                shD[(rb + g) * LDD + c] = fmaxf(acc[nt][g] + bv, 0.0f);
            }
        }
    }

    __syncthreads();

    // ---- Coalesced output: 16 lanes cover 256 contiguous bytes, NT stores ----
    {
        const int c4 = tid & 15;        // float4 column
        const int r0 = tid >> 4;        // rows 0..15, step 16
        #pragma unroll
        for (int j = 0; j < 8; ++j) {
            int rr = r0 + j * 16;
            int e  = e0 + rr;
            if (e < E) {
                vf4 v = *(const vf4*)&shD[rr * LDD + c4 * 4];
                __builtin_nontemporal_store(v, (vf4*)(out + (size_t)e * D_OUT + c4 * 4));
            }
        }
    }
}

extern "C" void kernel_launch(void* const* d_in, const int* in_sizes, int n_in,
                              void* d_out, int out_size, void* d_ws, size_t ws_size,
                              hipStream_t stream) {
    const float* node_feat = (const float*)d_in[0];
    const float* edge_feat = (const float*)d_in[1];
    const int*   src_idx   = (const int*)d_in[2];
    const int*   dst_idx   = (const int*)d_in[3];
    const float* W         = (const float*)d_in[4];
    const float* bias      = (const float*)d_in[5];
    float*       out       = (float*)d_out;
    unsigned*    wpk       = (unsigned*)d_ws;     // 24576 B packed-W image

    const int E = in_sizes[2];                 // number of edges

    pack_w_frags<<<(WPK_U32 + 255) / 256, 256, 0, stream>>>(W, wpk);

    const int grid = (E + TILE_M - 1) / TILE_M;
    edge_apply_wmma_bf16<<<grid, 256, 0, stream>>>(
        node_feat, edge_feat, src_idx, dst_idx, wpk, bias, out, E);
}
